// SAGE_996432413260
// MI455X (gfx1250) — matrix operand order repaired and checked by reading, not compiled, not run on hardware
//
#include <hip/hip_runtime.h>

// ---------------------------------------------------------------------------
// GraphSAGE (mean) 3-layer forward for MI455X / gfx1250.
// Aggregation: fp32 atomics (memory-bound, ~5GB traffic, L2-resident).
// GEMMs: v_wmma_f32_16x16x32_f16 (f16 operands, f32 accumulate).
// ---------------------------------------------------------------------------

typedef __attribute__((ext_vector_type(16))) _Float16 v16h;
typedef __attribute__((ext_vector_type(8)))  _Float16 v8h;
typedef __attribute__((ext_vector_type(8)))  float    v8f;

#define N_NODES    50000
#define F          256     // K dim of every layer input
#define LDS_STRIDE 264     // halves per LDS row (256 + 8 pad -> bank spread)

__global__ void zero_f32(float* __restrict__ p, int n) {
  int i = blockIdx.x * blockDim.x + threadIdx.x;
  if (i < n) p[i] = 0.0f;
}

__global__ void cvt_f32_to_f16(const float* __restrict__ w, _Float16* __restrict__ o, int n) {
  int i = blockIdx.x * blockDim.x + threadIdx.x;
  if (i < n) o[i] = (_Float16)w[i];
}

__global__ void deg_count(const int* __restrict__ dst, float* __restrict__ deg, int e) {
  int i = blockIdx.x * blockDim.x + threadIdx.x;
  if (i < e) atomicAdd(&deg[dst[i]], 1.0f);
}

// One thread per (edge, 4-feature chunk): gather float4 from src row,
// atomically accumulate into dst row of agg.
__global__ void scatter_add(const float* __restrict__ h, const int* __restrict__ src,
                            const int* __restrict__ dst, float* __restrict__ agg, int e) {
  int tid = blockIdx.x * blockDim.x + threadIdx.x;
  int ed = tid >> 6;              // 64 chunks of 4 floats per edge (F=256)
  if (ed >= e) return;
  int c = (tid & 63) << 2;
  int s = src[ed], d = dst[ed];
  const float4 v = *(const float4*)(h + (size_t)s * F + c);
  float* ap = agg + (size_t)d * F + c;
  atomicAdd(ap + 0, v.x);
  atomicAdd(ap + 1, v.y);
  atomicAdd(ap + 2, v.z);
  atomicAdd(ap + 3, v.w);
}

// A-fragment (contiguous k order, same as B loads): lanes 0-15 take K {k..k+15}
// of row (lane&15); lanes 16-31 take K {k+16..k+31}. Caller passes
// k0 = kk + (lane<16 ? 0 : 16); halves [0..7] from k0, [8..15] from k0+8.
__device__ __forceinline__ v16h lds_frag(const _Float16* base, int row, int k0) {
  const v8h lo = *(const v8h*)(base + row * LDS_STRIDE + k0);
  const v8h hi = *(const v8h*)(base + row * LDS_STRIDE + k0 + 8);
  return __builtin_shufflevector(lo, hi, 0, 1, 2, 3, 4, 5, 6, 7,
                                         8, 9, 10, 11, 12, 13, 14, 15);
}

// Fused: out[m, o] = act( sum_k h[m,k]*Ws[o,k] + (agg[m,k]/deg[m])*Wn[o,k] + b[o] )
// Block: 256 threads = 8 waves; block owns 16 rows (blockIdx.x) and 128 cols
// (blockIdx.y), one 16x16 tile per wave.
__global__ __launch_bounds__(256) void sage_gemm(
    const float* __restrict__ h, const float* __restrict__ agg,
    const float* __restrict__ deg,
    const _Float16* __restrict__ Ws, const _Float16* __restrict__ Wn,
    const float* __restrict__ bias, float* __restrict__ out,
    int n_out, int relu) {
  __shared__ _Float16 As[16 * LDS_STRIDE];   // self features, f16
  __shared__ _Float16 An[16 * LDS_STRIDE];   // mean-neighbor features, f16

  const int t = threadIdx.x;
  const int m_base = blockIdx.x * 16;

  // --- cooperative stage: 16 rows x 256 f32 -> f16 LDS, agg scaled by 1/deg ---
  {
    const int row = t >> 4;              // 0..15
    const int c   = (t & 15) * 16;       // 16 halves per thread
    const size_t g = (size_t)(m_base + row) * F + c;
    const float s = 1.0f / fmaxf(deg[m_base + row], 1.0f);
    v8h h0, h1, n0, n1;
#pragma unroll
    for (int j = 0; j < 8; ++j) {
      h0[j] = (_Float16)h[g + j];
      h1[j] = (_Float16)h[g + 8 + j];
      n0[j] = (_Float16)(agg[g + j] * s);
      n1[j] = (_Float16)(agg[g + 8 + j] * s);
    }
    *(v8h*)(As + row * LDS_STRIDE + c)     = h0;
    *(v8h*)(As + row * LDS_STRIDE + c + 8) = h1;
    *(v8h*)(An + row * LDS_STRIDE + c)     = n0;
    *(v8h*)(An + row * LDS_STRIDE + c + 8) = n1;
  }
  __syncthreads();

  // --- per-wave 16x16 WMMA tile ---
  const int wid  = t >> 5;                       // wave 0..7
  const int lane = t & 31;
  const int nt   = blockIdx.y * 8 + wid;         // column tile index
  const int col  = (nt << 4) + (lane & 15);      // B/C column for this lane
  const int row  = lane & 15;                    // A row for this lane
  const int ksel_a = (lane < 16) ? 0 : 16;       // A-frag K select
  const int ksel_b = (lane < 16) ? 0 : 16;       // B-frag K select

  const _Float16* wsrow = Ws + (size_t)col * F;  // W row o == B^T column o
  const _Float16* wnrow = Wn + (size_t)col * F;

  v8f acc = {};
#pragma unroll
  for (int kk = 0; kk < F; kk += 32) {           // self path
    v16h a = lds_frag(As, row, kk + ksel_a);
    v16h b = *(const v16h*)(wsrow + kk + ksel_b);
    acc = __builtin_amdgcn_wmma_f32_16x16x32_f16(false, a, false, b,
                                                 (short)0, acc, false, false);
  }
#pragma unroll
  for (int kk = 0; kk < F; kk += 32) {           // neighbor path
    v16h a = lds_frag(An, row, kk + ksel_a);
    v16h b = *(const v16h*)(wnrow + kk + ksel_b);
    acc = __builtin_amdgcn_wmma_f32_16x16x32_f16(false, a, false, b,
                                                 (short)0, acc, false, false);
  }

  // C/D layout: VGPR r -> M = r + (lane<16 ? 0 : 8); N = lane&15.
  const float bz = bias[col];
  const int r0 = m_base + ((lane < 16) ? 0 : 8);
#pragma unroll
  for (int r = 0; r < 8; ++r) {
    float v = acc[r] + bz;
    if (relu) v = fmaxf(v, 0.0f);
    out[(size_t)(r0 + r) * n_out + col] = v;
  }
}

extern "C" void kernel_launch(void* const* d_in, const int* in_sizes, int n_in,
                              void* d_out, int out_size, void* d_ws, size_t ws_size,
                              hipStream_t stream) {
  (void)n_in; (void)out_size; (void)ws_size;
  // setup_inputs() order: x, src, dst, Ws0, Wn0, b0, Ws1, Wn1, b1, Ws2, Wn2, b2
  const float* x   = (const float*)d_in[0];
  const int*   src = (const int*)d_in[1];
  const int*   dst = (const int*)d_in[2];
  const float* Wf[6]   = { (const float*)d_in[3], (const float*)d_in[4],
                           (const float*)d_in[6], (const float*)d_in[7],
                           (const float*)d_in[9], (const float*)d_in[10] };
  const float* bias[3] = { (const float*)d_in[5], (const float*)d_in[8],
                           (const float*)d_in[11] };
  const int E = in_sizes[1];

  // workspace carve-up (~155 MB)
  char* ws = (char*)d_ws;
  size_t o = 0;
  float* agg = (float*)(ws + o); o += (size_t)N_NODES * F * sizeof(float);
  float* hA  = (float*)(ws + o); o += (size_t)N_NODES * F * sizeof(float);
  float* hB  = (float*)(ws + o); o += (size_t)N_NODES * F * sizeof(float);
  float* deg = (float*)(ws + o); o += (size_t)N_NODES * sizeof(float);
  _Float16* w16base = (_Float16*)(ws + o);

  const int wsz[6] = { 256 * 256, 256 * 256, 256 * 256, 256 * 256, 128 * 256, 128 * 256 };
  _Float16* W16[6];
  { size_t wo = 0;
    for (int i = 0; i < 6; ++i) { W16[i] = w16base + wo; wo += (size_t)wsz[i]; } }

  // in-degrees (shared by all layers) + f16 weight conversion
  zero_f32<<<(N_NODES + 255) / 256, 256, 0, stream>>>(deg, N_NODES);
  deg_count<<<(E + 255) / 256, 256, 0, stream>>>(dst, deg, E);
  for (int i = 0; i < 6; ++i)
    cvt_f32_to_f16<<<(wsz[i] + 255) / 256, 256, 0, stream>>>(Wf[i], W16[i], wsz[i]);

  const float* hin = x;
  float*    houts[3] = { hA, hB, (float*)d_out };
  const int nouts[3] = { 256, 256, 128 };
  const int nAgg = N_NODES * F;

  for (int l = 0; l < 3; ++l) {
    zero_f32<<<(nAgg + 255) / 256, 256, 0, stream>>>(agg, nAgg);
    scatter_add<<<(int)(((size_t)E * 64 + 255) / 256), 256, 0, stream>>>(hin, src, dst, agg, E);
    dim3 grid(N_NODES / 16, nouts[l] / 128);   // 16 rows/block, 128 cols/block
    sage_gemm<<<grid, 256, 0, stream>>>(hin, agg, deg, W16[2 * l], W16[2 * l + 1],
                                        bias[l], houts[l], nouts[l], (l != 2) ? 1 : 0);
    hin = houts[l];
  }
}
